// FactorAtt_Mix_ConvRelPosEnc_12575664243016
// MI455X (gfx1250) — matrix operand order repaired and checked by reading, not compiled, not run on hardware
//
#include <hip/hip_runtime.h>

// ---------------------------------------------------------------------------
// MI455X (gfx1250, wave32) implementation of FactorAtt + ConvRelPosEnc.
// All matmuls run on v_wmma_f32_16x16x32_bf16 (bf16 A/B, f32 accumulate).
// Fragment gathers are expressed as uint4 LDS loads so they lower to
// ds_load_b128.
// ---------------------------------------------------------------------------

typedef __attribute__((ext_vector_type(16))) __bf16 v16bf;
typedef __attribute__((ext_vector_type(8)))  float  v8f;

#define NB    8
#define NSEQ  4096
#define NROWS (NB * NSEQ)   // 32768
#define CDIM  512
#define HEADS 8
#define CH    64

__device__ __forceinline__ unsigned short f2bf(float f) {
  union { float f; unsigned u; } x; x.f = f;
  unsigned r = x.u + 0x7FFFu + ((x.u >> 16) & 1u);   // round-to-nearest-even
  return (unsigned short)(r >> 16);
}
__device__ __forceinline__ float bf2f(unsigned short h) {
  union { unsigned u; float f; } x; x.u = ((unsigned)h) << 16;
  return x.f;
}

union FragB { v16bf v; unsigned short u[16]; unsigned u32[8]; uint4 q[2]; };
union FragF { v8f v; float f[8]; };

// ---------------------------------------------------------------------------
// f32 -> bf16 conversion
// ---------------------------------------------------------------------------
__global__ void cvt_bf16_kernel(const float* __restrict__ in,
                                unsigned short* __restrict__ out, int n) {
  int i = blockIdx.x * blockDim.x + threadIdx.x;
  if (i < n) out[i] = f2bf(in[i]);
}

// ---------------------------------------------------------------------------
// Generic strided GEMM:  C[M,Nc] = A[M,K] @ W[Nc,K]^T (+bias)
// A, W bf16 (as ushort); C f32 or bf16. Block = 64x64 tile, 256 thr / 8 waves,
// each wave owns two 16x16 WMMA tiles. K staged in 32-chunks through LDS.
// Requires 64|M, 64|Nc, 32|K, 8 | (lda, ldw, all base offsets).
//
// Fragment layouts used here (contiguous k order, wave32):
//  A 16x32 bf16 : lane = M | (khalf<<4); VGPR pairs cover K runs
//                 [kbA, kbA+15], kbA = (lane<16)?0:16
//  B 32x16 bf16 : lane = N | (khalf<<4); VGPR pairs cover K run
//                 [kbB, kbB+15], kbB = (lane<16)?0:16
//  C/D 16x16 f32: VGPR v -> row v + 8*(lane>=16), col = lane&15
// ---------------------------------------------------------------------------
__global__ __launch_bounds__(256)
void gemm_bf16_kernel(const unsigned short* __restrict__ A, int lda,
                      const unsigned short* __restrict__ W, int ldw,
                      const float* __restrict__ bias,
                      void* __restrict__ Cout, int ldc, int out_bf16, int K)
{
  __shared__ __align__(16) unsigned short As[64 * 32];
  __shared__ __align__(16) unsigned short Ws[64 * 32];

  const int tid  = threadIdx.x;
  const int m0   = blockIdx.y * 64;
  const int n0   = blockIdx.x * 64;
  const int wv   = tid >> 5, lane = tid & 31;
  const int tm   = wv & 3;              // 16-row tile within block (M)
  const int tn0  = (wv >> 2) * 2;       // first of two 16-col tiles (N)
  const int lrow = tid >> 2;            // staging: row of tile
  const int lseg = (tid & 3) * 8;       // staging: 8-elem segment

  const int laneM = lane & 15;
  const int kbA = (lane < 16) ? 0 : 16;
  const int kbB = (lane < 16) ? 0 : 16;

  // Pre-resolved LDS fragment pointers (all 16B-aligned)
  const uint4* apA  = (const uint4*)&As[(tm * 16 + laneM) * 32 + kbA];
  const uint4* bpB0 = (const uint4*)&Ws[((tn0    ) * 16 + laneM) * 32 + kbB];
  const uint4* bpB1 = (const uint4*)&Ws[((tn0 + 1) * 16 + laneM) * 32 + kbB];

  FragF acc0, acc1;
#pragma unroll
  for (int i = 0; i < 8; i++) { acc0.f[i] = 0.f; acc1.f[i] = 0.f; }

  for (int k0 = 0; k0 < K; k0 += 32) {
    uint4 av  = *(const uint4*)(A + (size_t)(m0 + lrow) * lda + k0 + lseg);
    uint4 wvv = *(const uint4*)(W + (size_t)(n0 + lrow) * ldw + k0 + lseg);
    __syncthreads();                       // finish prior fragment reads
    *(uint4*)&As[lrow * 32 + lseg] = av;
    *(uint4*)&Ws[lrow * 32 + lseg] = wvv;
    __syncthreads();
    if (k0 + 32 < K) {                     // hint next K-tile toward the caches
      __builtin_prefetch(A + (size_t)(m0 + lrow) * lda + k0 + 32 + lseg, 0, 1);
      __builtin_prefetch(W + (size_t)(n0 + lrow) * ldw + k0 + 32 + lseg, 0, 1);
    }

    FragB af, bf0, bf1;
    af.q[0]  = apA[0];   af.q[1]  = apA[1];     // K run  kbA..kbA+15
    bf0.q[0] = bpB0[0];  bf0.q[1] = bpB0[1];    // K run  kbB..kbB+15
    bf1.q[0] = bpB1[0];  bf1.q[1] = bpB1[1];

    acc0.v = __builtin_amdgcn_wmma_f32_16x16x32_bf16(false, af.v, false, bf0.v,
                                                     (short)0, acc0.v, false, false);
    acc1.v = __builtin_amdgcn_wmma_f32_16x16x32_bf16(false, af.v, false, bf1.v,
                                                     (short)0, acc1.v, false, false);
  }

  // Epilogue
  const int rbase = m0 + tm * 16 + ((lane < 16) ? 0 : 8);
#pragma unroll
  for (int v = 0; v < 8; v++) {
    int row = rbase + v;
    int c0 = n0 + tn0 * 16 + laneM;
    int c1 = c0 + 16;
    float v0 = acc0.f[v], v1 = acc1.f[v];
    if (bias) { v0 += bias[c0]; v1 += bias[c1]; }
    if (out_bf16) {
      unsigned short* O = (unsigned short*)Cout;
      O[(size_t)row * ldc + c0] = f2bf(v0);
      O[(size_t)row * ldc + c1] = f2bf(v1);
    } else {
      float* O = (float*)Cout;
      O[(size_t)row * ldc + c0] = v0;
      O[(size_t)row * ldc + c1] = v1;
    }
  }
}

// ---------------------------------------------------------------------------
// Column softmax stats over sequence axis: per (b, col) max and sum(exp).
// ---------------------------------------------------------------------------
__global__ __launch_bounds__(256)
void softmax_stats_kernel(const unsigned short* __restrict__ qkvb,
                          float* __restrict__ colmax, float* __restrict__ colsum)
{
  __shared__ float red[256];
  __shared__ float smax[64];
  const int t  = threadIdx.x;
  const int b  = blockIdx.x >> 3;
  const int cg = blockIdx.x & 7;
  const int cl = t & 63;                 // local col
  const int r0 = t >> 6;                 // 0..3
  const unsigned short* base = qkvb + (size_t)b * NSEQ * 1536 + 512 + cg * 64 + cl;

  float m = -1e30f;
  for (int n = r0; n < NSEQ; n += 4)
    m = fmaxf(m, bf2f(base[(size_t)n * 1536]));
  red[t] = m; __syncthreads();
  if (t < 64) {
    float mm = fmaxf(fmaxf(red[t], red[t + 64]), fmaxf(red[t + 128], red[t + 192]));
    smax[t] = mm;
    colmax[b * CDIM + cg * 64 + t] = mm;
  }
  __syncthreads();
  float mm = smax[cl];
  float s = 0.f;
  for (int n = r0; n < NSEQ; n += 4)
    s += __expf(bf2f(base[(size_t)n * 1536]) - mm);
  __syncthreads();
  red[t] = s; __syncthreads();
  if (t < 64)
    colsum[b * CDIM + cg * 64 + t] = red[t] + red[t + 64] + red[t + 128] + red[t + 192];
}

// ---------------------------------------------------------------------------
// kv^T per (b,h):  kvT[v,k] = attn_scale/sum_k * sum_n exp(K[n,k]-max_k)*V[n,v]
// LDS tiles stored TRANSPOSED (EsT[k][n], VsT[v][n]) so that both the A
// fragment (pairs over n at fixed k) and B fragments (pairs over n at fixed v)
// are contiguous 16B runs -> ds_load_b128, same pattern as the main GEMM.
// ---------------------------------------------------------------------------
__global__ __launch_bounds__(256)
void kv_wmma_kernel(const unsigned short* __restrict__ qkvb,
                    const float* __restrict__ colmax,
                    const float* __restrict__ colsum,
                    unsigned short* __restrict__ kvT, float attn_scale)
{
  __shared__ __align__(16) unsigned short EsT[64 * 32];  // [k][n]
  __shared__ __align__(16) unsigned short VsT[64 * 32];  // [v][n]
  __shared__ float smax[64], ssum[64];

  const int t = threadIdx.x;
  const int b = blockIdx.x >> 3, h = blockIdx.x & 7;
  if (t < 64) {
    smax[t] = colmax[b * CDIM + h * 64 + t];
    ssum[t] = colsum[b * CDIM + h * 64 + t];
  }
  __syncthreads();

  const int wv = t >> 5, lane = t & 31;
  const int tm = wv & 3, tn0 = (wv >> 2) * 2;
  const int laneM = lane & 15;
  const int kbA = (lane < 16) ? 0 : 16, kbB = (lane < 16) ? 0 : 16;
  const int lr = t >> 3, ls = (t & 7) * 8;   // staging: row n=lr, col group ls

  const uint4* apA  = (const uint4*)&EsT[(tm * 16 + laneM) * 32 + kbA];
  const uint4* bpB0 = (const uint4*)&VsT[((tn0    ) * 16 + laneM) * 32 + kbB];
  const uint4* bpB1 = (const uint4*)&VsT[((tn0 + 1) * 16 + laneM) * 32 + kbB];

  FragF acc0, acc1;
#pragma unroll
  for (int i = 0; i < 8; i++) { acc0.f[i] = 0.f; acc1.f[i] = 0.f; }

  const size_t rowbase = (size_t)b * NSEQ * 1536;
  for (int n0 = 0; n0 < NSEQ; n0 += 32) {
    size_t roff = rowbase + (size_t)(n0 + lr) * 1536;
    uint4 kvec = *(const uint4*)(qkvb + roff + 512  + h * 64 + ls);
    uint4 vvec = *(const uint4*)(qkvb + roff + 1024 + h * 64 + ls);
    unsigned short kp[8], vp[8];
    *(uint4*)kp = kvec;
    *(uint4*)vp = vvec;
    __syncthreads();
#pragma unroll
    for (int j = 0; j < 8; j++) {
      EsT[(ls + j) * 32 + lr] = f2bf(__expf(bf2f(kp[j]) - smax[ls + j]));
      VsT[(ls + j) * 32 + lr] = vp[j];
    }
    __syncthreads();

    FragB af, bf0, bf1;
    af.q[0]  = apA[0];   af.q[1]  = apA[1];
    bf0.q[0] = bpB0[0];  bf0.q[1] = bpB0[1];
    bf1.q[0] = bpB1[0];  bf1.q[1] = bpB1[1];

    acc0.v = __builtin_amdgcn_wmma_f32_16x16x32_bf16(false, af.v, false, bf0.v,
                                                     (short)0, acc0.v, false, false);
    acc1.v = __builtin_amdgcn_wmma_f32_16x16x32_bf16(false, af.v, false, bf1.v,
                                                     (short)0, acc1.v, false, false);
  }

  const size_t obase = (size_t)(b * 8 + h) * 64 * 64;
#pragma unroll
  for (int v = 0; v < 8; v++) {
    int kc = tm * 16 + v + ((lane < 16) ? 0 : 8);
    float sc = attn_scale / ssum[kc];
    int vc0 = tn0 * 16 + laneM, vc1 = vc0 + 16;
    kvT[obase + (size_t)vc0 * 64 + kc] = f2bf(acc0.f[v] * sc);
    kvT[obase + (size_t)vc1 * 64 + kc] = f2bf(acc1.f[v] * sc);
  }
}

// ---------------------------------------------------------------------------
// cat[row, h*128 + j] = j<64 ? mq[row,h,j] : q[row,h,j-64]   (bf16 copies)
// ---------------------------------------------------------------------------
__global__ void cat_pack_kernel(const unsigned short* __restrict__ mqb,
                                const unsigned short* __restrict__ qkvb,
                                unsigned short* __restrict__ cat)
{
  size_t i = (size_t)blockIdx.x * 256 + threadIdx.x;
  if (i >= (size_t)NROWS * 1024) return;
  int c = (int)(i & 1023);
  size_t row = i >> 10;
  int h = c >> 7, j = c & 127;
  cat[i] = (j < 64) ? mqb[row * 512 + h * 64 + j]
                    : qkvb[row * 1536 + h * 64 + (j - 64)];
}

// ---------------------------------------------------------------------------
// v (bf16, [b,n,ch] inside qkv) -> image layout f32 [b,ch,n], tiled transpose
// ---------------------------------------------------------------------------
__global__ __launch_bounds__(256)
void vimg_transpose_kernel(const unsigned short* __restrict__ qkvb,
                           float* __restrict__ vimg)
{
  __shared__ float tile[32][33];
  int tx = threadIdx.x, ty = threadIdx.y;
  int ch0 = blockIdx.x * 32, n0 = blockIdx.y * 32, b = blockIdx.z;
  const unsigned short* src = qkvb + (size_t)b * NSEQ * 1536 + 1024;
#pragma unroll
  for (int i = 0; i < 4; i++)
    tile[ty + 8 * i][tx] = bf2f(src[(size_t)(n0 + ty + 8 * i) * 1536 + ch0 + tx]);
  __syncthreads();
  float* dst = vimg + (size_t)b * CDIM * NSEQ;
#pragma unroll
  for (int i = 0; i < 4; i++)
    dst[(size_t)(ch0 + ty + 8 * i) * NSEQ + n0 + tx] = tile[tx][ty + 8 * i];
}

// ---------------------------------------------------------------------------
// Depthwise conv (3/5/7 per channel group), 64x64 image + halo in LDS
// ---------------------------------------------------------------------------
__global__ __launch_bounds__(256)
void dwconv_kernel(const float* __restrict__ vimg,
                   const float* __restrict__ w3, const float* __restrict__ b3,
                   const float* __restrict__ w5, const float* __restrict__ b5,
                   const float* __restrict__ w7, const float* __restrict__ b7,
                   float* __restrict__ cimg)
{
  __shared__ float tile[70 * 70];
  const int t  = threadIdx.x;
  const int b  = blockIdx.x >> 9, ch = blockIdx.x & 511;
  int ks; const float* wp; float bias;
  if (ch < 128)      { ks = 3; wp = w3 + ch * 9;          bias = b3[ch]; }
  else if (ch < 320) { ks = 5; wp = w5 + (ch - 128) * 25; bias = b5[ch - 128]; }
  else               { ks = 7; wp = w7 + (ch - 320) * 49; bias = b7[ch - 320]; }
  const int R = ks >> 1;
  const float* img = vimg + ((size_t)b * CDIM + ch) * NSEQ;
  for (int idx = t; idx < 70 * 70; idx += 256) {
    int yy = idx / 70, xx = idx - yy * 70;
    int gy = yy - 3, gx = xx - 3;
    float v = 0.f;
    if (gy >= 0 && gy < 64 && gx >= 0 && gx < 64) v = img[gy * 64 + gx];
    tile[idx] = v;
  }
  __syncthreads();
  float* out = cimg + ((size_t)b * CDIM + ch) * NSEQ;
  for (int p = t; p < 4096; p += 256) {
    int y = p >> 6, x = p & 63;
    float acc = bias;
    for (int dy = 0; dy < ks; dy++)
      for (int dx = 0; dx < ks; dx++)
        acc += wp[dy * ks + dx] * tile[(y + dy - R + 3) * 70 + (x + dx - R + 3)];
    out[p] = acc;
  }
}

// conv image [b,ch,n] -> [b,n,ch]
__global__ __launch_bounds__(256)
void convT_transpose_kernel(const float* __restrict__ cimg,
                            float* __restrict__ convT)
{
  __shared__ float tile[32][33];
  int tx = threadIdx.x, ty = threadIdx.y;
  int n0 = blockIdx.x * 32, ch0 = blockIdx.y * 32, b = blockIdx.z;
  const float* src = cimg + (size_t)b * CDIM * NSEQ;
#pragma unroll
  for (int i = 0; i < 4; i++)
    tile[ty + 8 * i][tx] = src[(size_t)(ch0 + ty + 8 * i) * NSEQ + n0 + tx];
  __syncthreads();
  float* dst = convT + (size_t)b * NSEQ * CDIM;
#pragma unroll
  for (int i = 0; i < 4; i++)
    dst[(size_t)(n0 + ty + 8 * i) * CDIM + ch0 + tx] = tile[tx][ty + 8 * i];
}

// out = scale*factor_att + q_mix * conv_v  -> bf16 (input to projection GEMM)
__global__ void combine_kernel(const float* __restrict__ fattn,
                               const float* __restrict__ qmix,
                               const float* __restrict__ convT,
                               unsigned short* __restrict__ attnb)
{
  size_t i = (size_t)blockIdx.x * 256 + threadIdx.x;
  if (i >= (size_t)NROWS * CDIM) return;
  attnb[i] = f2bf(fattn[i] + qmix[i] * convT[i]);
}

// ---------------------------------------------------------------------------
extern "C" void kernel_launch(void* const* d_in, const int* in_sizes, int n_in,
                              void* d_out, int out_size, void* d_ws, size_t ws_size,
                              hipStream_t stream)
{
  const float* x     = (const float*)d_in[0];
  const float* mx    = (const float*)d_in[1];
  const float* Wqkv  = (const float*)d_in[2];
  const float* Wproj = (const float*)d_in[3];
  const float* bproj = (const float*)d_in[4];
  const float* Wmix  = (const float*)d_in[5];
  const float* bmix  = (const float*)d_in[6];
  const float* cw3   = (const float*)d_in[7];
  const float* cb3   = (const float*)d_in[8];
  const float* cw5   = (const float*)d_in[9];
  const float* cb5   = (const float*)d_in[10];
  const float* cw7   = (const float*)d_in[11];
  const float* cb7   = (const float*)d_in[12];

  char* ws = (char*)d_ws;
  size_t off = 0;
  auto alloc = [&](size_t bytes) -> char* {
    char* p = ws + off;
    off = (off + bytes + 255) & ~(size_t)255;
    return p;
  };
  unsigned short* xb    = (unsigned short*)alloc((size_t)NROWS * 512 * 2);   // -> attnb
  unsigned short* mxb   = (unsigned short*)alloc((size_t)NROWS * 512 * 2);
  unsigned short* qkvb  = (unsigned short*)alloc((size_t)NROWS * 1536 * 2);
  unsigned short* mqb   = (unsigned short*)alloc((size_t)NROWS * 512 * 2);
  unsigned short* catb  = (unsigned short*)alloc((size_t)NROWS * 1024 * 2);  // -> fattn
  float*          qmix  = (float*)alloc((size_t)NROWS * 512 * 4);
  float*          vimg  = (float*)alloc((size_t)NB * CDIM * NSEQ * 4);       // -> convT
  float*          cimg  = (float*)alloc((size_t)NB * CDIM * NSEQ * 4);
  unsigned short* Wqkvb = (unsigned short*)alloc((size_t)1536 * 512 * 2);
  unsigned short* Wprojb= (unsigned short*)alloc((size_t)512 * 512 * 2);
  unsigned short* Wmixb = (unsigned short*)alloc((size_t)64 * 128 * 2);
  unsigned short* kvT   = (unsigned short*)alloc((size_t)NB * HEADS * 64 * 64 * 2);
  float*          colmax= (float*)alloc((size_t)NB * CDIM * 4);
  float*          colsum= (float*)alloc((size_t)NB * CDIM * 4);
  float*          fattn = (float*)catb;   // cat dead before factor_att writes
  float*          convT = vimg;           // vimg dead before transpose-back
  unsigned short* attnb = xb;             // xb dead before combine writes

  auto cvt = [&](const float* in, unsigned short* out, size_t n) {
    cvt_bf16_kernel<<<(unsigned)((n + 255) / 256), 256, 0, stream>>>(in, out, (int)n);
  };
  cvt(x,     xb,    (size_t)NROWS * 512);
  cvt(mx,    mxb,   (size_t)NROWS * 512);
  cvt(Wqkv,  Wqkvb, (size_t)1536 * 512);
  cvt(Wproj, Wprojb,(size_t)512 * 512);
  cvt(Wmix,  Wmixb, (size_t)64 * 128);

  // qkv = x @ Wqkv^T  (bf16 out)
  gemm_bf16_kernel<<<dim3(1536 / 64, NROWS / 64), 256, 0, stream>>>(
      xb, 512, Wqkvb, 512, nullptr, (void*)qkvb, 1536, 1, 512);
  // mq = mx @ Wqkv[:512]^T  (bf16 out)
  gemm_bf16_kernel<<<dim3(512 / 64, NROWS / 64), 256, 0, stream>>>(
      mxb, 512, Wqkvb, 512, nullptr, (void*)mqb, 512, 1, 512);

  // cat = [mq, q] per head
  cat_pack_kernel<<<(unsigned)(((size_t)NROWS * 1024) / 256), 256, 0, stream>>>(
      mqb, qkvb, catb);

  // q_mix (per head): cat_h @ Wmix^T + bmix
  for (int h = 0; h < HEADS; h++)
    gemm_bf16_kernel<<<dim3(1, NROWS / 64), 256, 0, stream>>>(
        catb + h * 128, 1024, Wmixb, 128, bmix,
        (void*)(qmix + h * 64), 512, 0, 128);

  // softmax stats over sequence dim of k
  softmax_stats_kernel<<<NB * 8, 256, 0, stream>>>(qkvb, colmax, colsum);

  // kvT (scaled, normalized) per (b,h)
  kv_wmma_kernel<<<NB * HEADS, 256, 0, stream>>>(qkvb, colmax, colsum, kvT, 0.125f);

  // factor_att = mq @ kv  (scale folded into kvT), per (b,h)
  for (int bh = 0; bh < NB * HEADS; bh++) {
    int b = bh >> 3, h = bh & 7;
    gemm_bf16_kernel<<<dim3(1, NSEQ / 64), 256, 0, stream>>>(
        mqb + (size_t)b * NSEQ * 512 + h * 64, 512,
        kvT + (size_t)bh * 64 * 64, 64, nullptr,
        (void*)(fattn + (size_t)b * NSEQ * 512 + h * 64), 512, 0, 64);
  }

  // depthwise conv path
  vimg_transpose_kernel<<<dim3(16, 128, NB), dim3(32, 8), 0, stream>>>(qkvb, vimg);
  dwconv_kernel<<<NB * CDIM, 256, 0, stream>>>(vimg, cw3, cb3, cw5, cb5, cw7, cb7, cimg);
  convT_transpose_kernel<<<dim3(128, 16, NB), dim3(32, 8), 0, stream>>>(cimg, convT);

  // combine -> bf16
  combine_kernel<<<(unsigned)(((size_t)NROWS * CDIM) / 256), 256, 0, stream>>>(
      fattn, qmix, convT, attnb);

  // out = attn @ Wproj^T + bproj  (f32 out)
  gemm_bf16_kernel<<<dim3(512 / 64, NROWS / 64), 256, 0, stream>>>(
      attnb, 512, Wprojb, 512, bproj, d_out, 512, 0, 512);
}
